// DSGATA1_83330955477490
// MI455X (gfx1250) — matrix operand order repaired and checked by reading, not compiled, not run on hardware
//
#include <hip/hip_runtime.h>

// ---------------------------------------------------------------------------
// CDNA5 (gfx1250) wave32 WMMA types
// ---------------------------------------------------------------------------
typedef __attribute__((ext_vector_type(16))) __bf16 v16bf;
typedef __attribute__((ext_vector_type(8)))  float  v8f;

union FragA { unsigned int u[8]; v16bf v; };

__device__ __forceinline__ unsigned short f2bf_bits(float f) {
  // round-to-nearest-even f32 -> bf16 (bit pattern)
  unsigned int u = __float_as_uint(f);
  u += 0x7FFFu + ((u >> 16) & 1u);
  return (unsigned short)(u >> 16);
}

__device__ __forceinline__ v8f wmma_bf16(v16bf a, v16bf b, v8f c) {
  // 8 args: (neg_a, A, neg_b, B, c_mod, C, reuse_a, reuse_b)
  return __builtin_amdgcn_wmma_f32_16x16x32_bf16(false, a, false, b, (short)0, c,
                                                 false, false);
}

// Async copy of 16 bytes global -> LDS (GLOBAL_LOAD_ASYNC_TO_LDS_B128,
// ASYNCcnt-tracked).  lds_ptr: generic pointer to __shared__; low 32 bits are
// the wave-relative LDS byte address (CDNA5 flat->LDS truncation rule).
__device__ __forceinline__ void async_ld_b128(void* lds_ptr, const void* gptr) {
  unsigned lds = (unsigned)(unsigned long long)lds_ptr;
  asm volatile("global_load_async_to_lds_b128 %0, %1, off"
               :: "v"(lds), "v"(gptr) : "memory");
}
__device__ __forceinline__ void async_wait0() {
  asm volatile("s_wait_asynccnt 0x0" ::: "memory");
}

// A-matrix fragment (16x32 bf16, MxK) from LDS, row-major rows of `rowStride`
// bf16 elements starting at column k0.  Contiguous k order (this code's choice):
//   lane<16 : row M=lane,    K in {0..15}
//   lane>=16: row M=lane-16, K in {16..31}
// K values are packed in consecutive pairs -> read as 32-bit words (merges to
// ds_load_b128).
__device__ __forceinline__ v16bf load_a_frag(const unsigned short* sA,
                                             int rowStride, int k0) {
  int lane = threadIdx.x & 31;
  int row  = lane & 15;
  int hb   = (lane >> 4) * 8;  // 0 or 8
  FragA f;
#pragma unroll
  for (int p = 0; p < 8; ++p) {
    int k = 2 * hb + 2 * p;
    f.u[p] = *(const unsigned int*)(sA + row * rowStride + k0 + k);
  }
  return f.v;
}

// ---------------------------------------------------------------------------
// Weight packing: f32 KxN row-major -> bf16 WMMA B-fragments (32x16, KxN).
// lane<16 holds col N=lane, K=0..15; lane>=16 holds col N=lane-16, K=16..31;
// fragment contiguous per lane: one 32 B vector load at GEMM time.
// ---------------------------------------------------------------------------
__global__ __launch_bounds__(32)
void pack_b_bf16(const float* __restrict__ W, unsigned short* __restrict__ P,
                 int K, int Nw) {
  int Nt   = Nw >> 4;
  int tile = blockIdx.x;            // tile = kt*Nt + nt
  int kt   = tile / Nt;
  int nt   = tile - kt * Nt;
  int lane = threadIdx.x;
  int n    = lane & 15;
  int hb   = (lane >> 4) * 16;
#pragma unroll
  for (int j = 0; j < 16; ++j) {
    int k = hb + j;
    float v = W[(long long)(kt * 32 + k) * Nw + nt * 16 + n];
    P[(long long)tile * 512 + lane * 16 + j] = f2bf_bits(v);
  }
}

// ---------------------------------------------------------------------------
// x(bf16) = node_features[entity] @ W_proj + b    [N,128]@[128,64]
// (gathered f32 rows -> bf16 in LDS; rows assumed multiple of 16)
// ---------------------------------------------------------------------------
__global__ __launch_bounds__(128)
void k_proj(const int* __restrict__ entity, const float* __restrict__ nf,
            const unsigned short* __restrict__ Bp, const float* __restrict__ bias,
            unsigned short* __restrict__ C, int N) {
  __shared__ __align__(16) unsigned short sA[16 * 128];
  int tile = blockIdx.x, tid = threadIdx.x;
  for (int i = tid; i < 16 * 128; i += 128) {
    int r = i >> 7, c = i & 127;
    int node = entity[tile * 16 + r];
    sA[i] = f2bf_bits(nf[(long long)node * 128 + c]);
  }
  __syncthreads();
  int wave = tid >> 5, lane = tid & 31;
  v8f acc = {};
#pragma unroll
  for (int kt = 0; kt < 4; ++kt) {
    v16bf a = load_a_frag(sA, 128, kt * 32);
    v16bf b = ((const v16bf*)Bp)[(kt * 4 + wave) * 32 + lane];
    acc = wmma_bf16(a, b, acc);
  }
  int col = wave * 16 + (lane & 15);
  float bv = bias[col];
  int hb = (lane >> 4) * 8;
#pragma unroll
  for (int r = 0; r < 8; ++r) {
    int row = tile * 16 + r + hb;
    C[(long long)row * 64 + col] = f2bf_bits(acc[r] + bv);
  }
}

// ---------------------------------------------------------------------------
// Dense row-GEMM, bf16 activations in, f32 or bf16 out:
//   C[rows,NOUT] = A(bf16)[rows,KDIM] @ Bp + bias
// A tile (16 x KDIM bf16, contiguous) staged with async global->LDS b128.
// ---------------------------------------------------------------------------
template <int KDIM, int NOUT, int WAVES, bool OUTBF>
__global__ __launch_bounds__(WAVES * 32)
void k_gemm_bf16(const unsigned short* __restrict__ A,
                 const unsigned short* __restrict__ Bp,
                 const float* __restrict__ bias, void* __restrict__ Cout, int rows) {
  constexpr int NT     = NOUT / 16;
  constexpr int NT_W   = NT / WAVES;
  constexpr int KT     = KDIM / 32;
  constexpr int CHUNKS = 16 * KDIM * 2 / 16;   // 16B chunks in the A tile
  __shared__ __align__(16) unsigned short sA[16 * KDIM];
  int tile = blockIdx.x, tid = threadIdx.x;

  const char* gbase = (const char*)(A + (size_t)tile * 16 * KDIM);
  for (int i = tid; i < CHUNKS; i += WAVES * 32)
    async_ld_b128(&sA[i * 8], gbase + i * 16);
  async_wait0();
  __syncthreads();

  int wave = tid >> 5, lane = tid & 31;
  v8f zero = {};
  v8f acc[NT_W];
#pragma unroll
  for (int t = 0; t < NT_W; ++t) acc[t] = zero;
#pragma unroll
  for (int kt = 0; kt < KT; ++kt) {
    v16bf a = load_a_frag(sA, KDIM, kt * 32);
#pragma unroll
    for (int t = 0; t < NT_W; ++t) {
      int nt = wave * NT_W + t;
      v16bf b = ((const v16bf*)Bp)[(kt * NT + nt) * 32 + lane];
      acc[t] = wmma_bf16(a, b, acc[t]);
    }
  }
  int hb = (lane >> 4) * 8;
#pragma unroll
  for (int t = 0; t < NT_W; ++t) {
    int col = (wave * NT_W + t) * 16 + (lane & 15);
    float bv = bias[col];
#pragma unroll
    for (int r = 0; r < 8; ++r) {
      int row = tile * 16 + r + hb;
      float v = acc[t][r] + bv;
      if constexpr (OUTBF)
        ((unsigned short*)Cout)[(size_t)row * NOUT + col] = f2bf_bits(v);
      else
        ((float*)Cout)[(size_t)row * NOUT + col] = v;
    }
  }
}

// h_mean(bf16)[n,d] = mean over 4 heads of h(f32)[n, h*64+d]
__global__ void k_hmean(const float* __restrict__ h, unsigned short* __restrict__ hm,
                        int N) {
  int i = blockIdx.x * blockDim.x + threadIdx.x;
  if (i >= N * 64) return;
  int n = i >> 6, d = i & 63;
  const float* p = h + (long long)n * 256 + d;
  hm[i] = f2bf_bits(0.25f * (p[0] + p[64] + p[128] + p[192]));
}

// ef(bf16)[e,:] = rel_emb[edge_type[e],:]
__global__ void k_ef0(const int* __restrict__ et, const float* __restrict__ rel,
                      unsigned short* __restrict__ ef, int E) {
  int i = blockIdx.x * blockDim.x + threadIdx.x;
  if (i >= E * 64) return;
  int e = i >> 6, d = i & 63;
  ef[i] = f2bf_bits(rel[(long long)et[e] * 64 + d]);
}

// ---------------------------------------------------------------------------
// Fused fc_edges: async-gather [h_mean[src] | efp | h_mean[dst]] (bf16) into
// LDS, GEMM 16x192 @ 192x256 (12 WMMAs), leaky-ReLU in registers, attention
// logits a[e,h]=<f_out,fa_W> and head-mean ef_out via LDS ds_add reductions
// (f_out is never materialized in HBM), segment-max via ordered-u32 atomics.
// ---------------------------------------------------------------------------
__global__ __launch_bounds__(256)
void k_edge_attn(const int* __restrict__ src, const int* __restrict__ dst,
                 const unsigned short* __restrict__ hmean,
                 const unsigned short* __restrict__ efp,
                 const unsigned short* __restrict__ Wfe,
                 const float* __restrict__ faW,
                 float* __restrict__ abuf, unsigned int* __restrict__ mord,
                 unsigned short* __restrict__ efout, int write_ef, int E) {
  __shared__ __align__(16) unsigned short sA[16 * 192];  // 16 x [hm_s|efp|hm_d]
  __shared__ float sEf[16 * 64];
  __shared__ float sAttn[16 * 4];
  __shared__ int   sSrc[16], sDst[16];
  int tile = blockIdx.x, tid = threadIdx.x;
  int e0 = tile * 16;

  if (tid < 16) { sSrc[tid] = src[e0 + tid]; sDst[tid] = dst[e0 + tid]; }
  for (int i = tid; i < 16 * 64; i += 256) sEf[i] = 0.f;
  if (tid < 64) sAttn[tid] = 0.f;
  __syncthreads();

  // 16 rows x 384 B, 24 x 16B chunks per row: async global->LDS gather
  for (int i = tid; i < 16 * 24; i += 256) {
    int row = i / 24, sub = i - row * 24;
    const void* g;
    if (sub < 8)       g = hmean + (size_t)sSrc[row] * 64 + sub * 8;
    else if (sub < 16) g = efp   + (size_t)(e0 + row) * 64 + (sub - 8) * 8;
    else               g = hmean + (size_t)sDst[row] * 64 + (sub - 16) * 8;
    async_ld_b128(&sA[i * 8], g);
  }
  async_wait0();
  __syncthreads();

  int wave = tid >> 5, lane = tid & 31;
  v8f acc0 = {}, acc1 = {};
  const v16bf* B = (const v16bf*)Wfe;
#pragma unroll
  for (int kt = 0; kt < 6; ++kt) {
    v16bf a = load_a_frag(sA, 192, kt * 32);
    if (kt < 5) __builtin_prefetch(&B[((kt + 1) * 16 + wave * 2) * 32 + lane], 0, 1);
    v16bf b0 = B[(kt * 16 + wave * 2 + 0) * 32 + lane];
    v16bf b1 = B[(kt * 16 + wave * 2 + 1) * 32 + lane];
    acc0 = wmma_bf16(a, b0, acc0);
    acc1 = wmma_bf16(a, b1, acc1);
  }

  int hb = (lane >> 4) * 8;
#pragma unroll
  for (int t = 0; t < 2; ++t) {
    v8f acc = t ? acc1 : acc0;
    int col  = (wave * 2 + t) * 16 + (lane & 15);
    int head = col >> 6, d = col & 63;
    float fa = faW[d];
#pragma unroll
    for (int r = 0; r < 8; ++r) {
      float c = acc[r];
      c = (c > 0.f) ? c : 0.01f * c;   // leaky_relu(0.01)
      int row = r + hb;
      atomicAdd(&sAttn[row * 4 + head], c * fa);
      if (write_ef) atomicAdd(&sEf[row * 64 + d], 0.25f * c);
    }
  }
  __syncthreads();

  if (tid < 64) {
    int row = tid >> 2, head = tid & 3;
    float aval = sAttn[tid];
    abuf[(long long)(e0 + row) * 4 + head] = aval;
    // order-preserving f32 -> u32 for atomic segment-max (init pattern 0x0)
    unsigned int u = __float_as_uint(aval);
    unsigned int ord = (u & 0x80000000u) ? ~u : (u | 0x80000000u);
    atomicMax(&mord[(long long)sDst[row] * 4 + head], ord);
  }
  if (write_ef) {
    for (int i = tid; i < 16 * 64; i += 256) {
      int row = i >> 6, d = i & 63;
      efout[(size_t)(e0 + row) * 64 + d] = f2bf_bits(sEf[i]);
    }
  }
}

// exa = exp(a - max[dst]); denom[dst] += exa
__global__ void k_softmax_norm(const int* __restrict__ dst, float* __restrict__ abuf,
                               const unsigned int* __restrict__ mord,
                               float* __restrict__ denom, int E) {
  int i = blockIdx.x * blockDim.x + threadIdx.x;
  if (i >= E * 4) return;
  int e = i >> 2, h = i & 3;
  int dn = dst[e];
  unsigned int ord = mord[(long long)dn * 4 + h];
  unsigned int u = (ord & 0x80000000u) ? (ord & 0x7FFFFFFFu) : ~ord;
  float m = __uint_as_float(u);
  float ex = expf(abuf[i] - m);
  abuf[i] = ex;
  atomicAdd(&denom[(long long)dn * 4 + h], ex);
}

// h_new[dst] += (exa/denom[dst]) * h[src]   (per head,d)
__global__ __launch_bounds__(256)
void k_aggregate(const int* __restrict__ src, const int* __restrict__ dst,
                 const float* __restrict__ abuf, const float* __restrict__ denom,
                 const float* __restrict__ hbuf, float* __restrict__ hnew, int E) {
  int e = blockIdx.x;
  int t = threadIdx.x;           // t = head*64 + d
  int h = t >> 6;
  int s = src[e], dn = dst[e];
  float coef = abuf[(long long)e * 4 + h] / denom[(long long)dn * 4 + h];
  atomicAdd(&hnew[(long long)dn * 256 + t], coef * hbuf[(long long)s * 256 + t]);
}

// x = mean over heads; ELU + bf16 out between layers, f32 out at the end
__global__ void k_node_out(const float* __restrict__ hnew,
                           unsigned short* __restrict__ xbf,
                           float* __restrict__ xf32, int N, int apply_elu) {
  int i = blockIdx.x * blockDim.x + threadIdx.x;
  if (i >= N * 64) return;
  int n = i >> 6, d = i & 63;
  const float* p = hnew + (long long)n * 256 + d;
  float v = 0.25f * (p[0] + p[64] + p[128] + p[192]);
  if (apply_elu && v < 0.f) v = expf(v) - 1.f;
  if (xf32) xf32[i] = v;
  else      xbf[i] = f2bf_bits(v);
}

// ---------------------------------------------------------------------------
// Host launcher
// ---------------------------------------------------------------------------
static inline size_t align256(size_t x) { return (x + 255) & ~(size_t)255; }

extern "C" void kernel_launch(void* const* d_in, const int* in_sizes, int n_in,
                              void* d_out, int out_size, void* d_ws, size_t ws_size,
                              hipStream_t stream) {
  if (n_in < 13) return;
  const int*   entity  = (const int*)d_in[0];
  const int*   eidx    = (const int*)d_in[1];
  const int*   etype   = (const int*)d_in[2];
  const float* nfeat   = (const float*)d_in[3];
  const float* W_proj  = (const float*)d_in[4];
  const float* b_proj  = (const float*)d_in[5];
  const float* rel_emb = (const float*)d_in[6];
  const float* ep_W    = (const float*)d_in[7];
  const float* ep_b    = (const float*)d_in[8];
  const float* fn_W    = (const float*)d_in[9];
  const float* fn_b    = (const float*)d_in[10];
  const float* fe_W    = (const float*)d_in[11];
  const float* fa_W    = (const float*)d_in[12];

  const int N = in_sizes[0];     // 40000  (multiple of 16)
  const int E = in_sizes[2];     // 400000 (multiple of 16)
  const int L = 2;
  const int* src = eidx;
  const int* dst = eidx + E;

  // workspace carve (activations feeding WMMA are bf16; f_out never stored)
  char* p = (char*)d_ws;
  auto carve = [&](size_t bytes) { char* r = p; p += align256(bytes); return r; };
  unsigned short* x_ws    = (unsigned short*)carve((size_t)N * 64 * 2);
  unsigned short* ef_ws   = (unsigned short*)carve((size_t)E * 64 * 2);
  unsigned short* efp_ws  = (unsigned short*)carve((size_t)E * 64 * 2);
  float*          h_ws    = (float*)carve((size_t)N * 256 * 4);
  unsigned short* hm_ws   = (unsigned short*)carve((size_t)N * 64 * 2);
  float*          a_ws    = (float*)carve((size_t)E * 4 * 4);
  unsigned int*   mord_ws = (unsigned int*)carve((size_t)N * 4 * 4);
  float*          den_ws  = (float*)carve((size_t)N * 4 * 4);
  float*          hnew_ws = (float*)carve((size_t)N * 256 * 4);
  unsigned short* pb_proj = (unsigned short*)carve((size_t)16 * 512 * 2);
  unsigned short* pb_ep   = (unsigned short*)carve((size_t)L * 8 * 512 * 2);
  unsigned short* pb_fn   = (unsigned short*)carve((size_t)L * 32 * 512 * 2);
  unsigned short* pb_fe   = (unsigned short*)carve((size_t)L * 96 * 512 * 2);

  // pack all weights to bf16 B-fragments (tiny, deterministic every call)
  pack_b_bf16<<<16, 32, 0, stream>>>(W_proj, pb_proj, 128, 64);
  for (int l = 0; l < L; ++l) {
    pack_b_bf16<<<8,  32, 0, stream>>>(ep_W + (size_t)l * 64 * 64,   pb_ep + (size_t)l * 8 * 512,  64, 64);
    pack_b_bf16<<<32, 32, 0, stream>>>(fn_W + (size_t)l * 64 * 256,  pb_fn + (size_t)l * 32 * 512, 64, 256);
    pack_b_bf16<<<96, 32, 0, stream>>>(fe_W + (size_t)l * 192 * 256, pb_fe + (size_t)l * 96 * 512, 192, 256);
  }

  // initial features
  k_ef0<<<(E * 64 + 255) / 256, 256, 0, stream>>>(etype, rel_emb, ef_ws, E);
  k_proj<<<N / 16, 128, 0, stream>>>(entity, nfeat, pb_proj, b_proj, x_ws, N);

  for (int l = 0; l < L; ++l) {
    // edge_project: efp(bf16) = ef @ ep_W[l] + ep_b[l]
    k_gemm_bf16<64, 64, 4, true><<<E / 16, 128, 0, stream>>>(
        ef_ws, pb_ep + (size_t)l * 8 * 512, ep_b + (size_t)l * 64, efp_ws, E);
    // fc_nodes: h(f32) = x @ fn_W[l] + fn_b[l]
    k_gemm_bf16<64, 256, 8, false><<<N / 16, 256, 0, stream>>>(
        x_ws, pb_fn + (size_t)l * 32 * 512, fn_b + (size_t)l * 256, h_ws, N);
    k_hmean<<<(N * 64 + 255) / 256, 256, 0, stream>>>(h_ws, hm_ws, N);

    hipMemsetAsync(mord_ws, 0, (size_t)N * 4 * 4, stream);
    hipMemsetAsync(den_ws,  0, (size_t)N * 4 * 4, stream);
    hipMemsetAsync(hnew_ws, 0, (size_t)N * 256 * 4, stream);

    // fused fc_edges + leaky_relu + attn logits + head-mean ef + segment max
    k_edge_attn<<<E / 16, 256, 0, stream>>>(
        src, dst, hm_ws, efp_ws, pb_fe + (size_t)l * 96 * 512,
        fa_W + (size_t)l * 64, a_ws, mord_ws, ef_ws, (l < L - 1) ? 1 : 0, E);
    // edge softmax normalization
    k_softmax_norm<<<(E * 4 + 255) / 256, 256, 0, stream>>>(dst, a_ws, mord_ws, den_ws, E);
    // weighted aggregation
    k_aggregate<<<E, 256, 0, stream>>>(src, dst, a_ws, den_ws, h_ws, hnew_ws, E);

    if (l == L - 1)
      k_node_out<<<(N * 64 + 255) / 256, 256, 0, stream>>>(hnew_ws, nullptr, (float*)d_out, N, 0);
    else
      k_node_out<<<(N * 64 + 255) / 256, 256, 0, stream>>>(hnew_ws, x_ws, nullptr, N, 1);
  }
}
